// CrossAttention_68564857913662
// MI455X (gfx1250) — matrix operand order repaired and checked by reading, not compiled, not run on hardware
//
#include <hip/hip_runtime.h>
#include <hip/hip_bf16.h>

typedef __bf16 bf16_t;
typedef __attribute__((ext_vector_type(16))) __bf16 v16bf;
typedef __attribute__((ext_vector_type(8)))  float  v8f;
typedef __attribute__((ext_vector_type(4)))  int    v4i;

#define HCONST 16
#define DH     64
#define CDIM   1024

__device__ __forceinline__ v8f wmma_bf16(v16bf a, v16bf b, v8f c) {
  // D = A(16x32 bf16) * B(32x16 bf16) + C(16x16 f32)
  return __builtin_amdgcn_wmma_f32_16x16x32_bf16(false, a, false, b, (short)0, c,
                                                 false, false);
}

// ---- CDNA5 async global->LDS (ASYNCcnt-tracked DMA), 16B per lane -----------
__device__ __forceinline__ void async_ld_b128(bf16_t* lds_dst, const bf16_t* gsrc) {
#if __has_builtin(__builtin_amdgcn_global_load_async_to_lds_b128)
  __builtin_amdgcn_global_load_async_to_lds_b128(
      (__attribute__((address_space(1))) v4i*)(uintptr_t)gsrc,
      (__attribute__((address_space(3))) v4i*)lds_dst,
      0, 0);
#else
  const unsigned loff =
      (unsigned)(uintptr_t)(__attribute__((address_space(3))) void*)lds_dst;
  asm volatile("global_load_async_to_lds_b128 %0, %1, off"
               :: "v"(loff), "v"((unsigned long long)(uintptr_t)gsrc)
               : "memory");
#endif
}

__device__ __forceinline__ void wait_async0() {
#if __has_builtin(__builtin_amdgcn_s_wait_asynccnt)
  __builtin_amdgcn_s_wait_asynccnt(0);
#else
  asm volatile("s_wait_asynccnt 0x0" ::: "memory");
#endif
}

// ---- XOR butterfly via DPP (no LDS round-trip, stays within 16-lane halves) -
template<int XM>
__device__ __forceinline__ float bfly_xor(float x) {
  const int xi = __builtin_bit_cast(int, x);
  int y;
  if constexpr (XM == 8) {
    y = __builtin_amdgcn_mov_dpp(xi, 0x128, 0xf, 0xf, true);   // row_ror:8
  } else {
    constexpr int sel = ((0 ^ XM) << 0)  | ((1 ^ XM) << 3)  | ((2 ^ XM) << 6)  |
                        ((3 ^ XM) << 9)  | ((4 ^ XM) << 12) | ((5 ^ XM) << 15) |
                        ((6 ^ XM) << 18) | ((7 ^ XM) << 21);
    y = __builtin_amdgcn_mov_dpp8(xi, sel);
  }
  return __builtin_bit_cast(float, y);
}

__device__ __forceinline__ float half_max16(float v) {
  v = fmaxf(v, bfly_xor<1>(v));
  v = fmaxf(v, bfly_xor<2>(v));
  v = fmaxf(v, bfly_xor<4>(v));
  v = fmaxf(v, bfly_xor<8>(v));
  return v;
}
__device__ __forceinline__ float half_sum16(float v) {
  v += bfly_xor<1>(v);
  v += bfly_xor<2>(v);
  v += bfly_xor<4>(v);
  v += bfly_xor<8>(v);
  return v;
}

// ---------------------------------------------------------------------------
// fp32 -> bf16 bulk convert (one pass; removes converts from GEMM hot loops)
// ---------------------------------------------------------------------------
__global__ void cvt_bf16_kernel(const float* __restrict__ src,
                                bf16_t* __restrict__ dst, int n)
{
  const int i = (blockIdx.x * blockDim.x + threadIdx.x) * 8;
  if (i + 8 > n) return;
#pragma unroll
  for (int e = 0; e < 8; ++e) dst[i + e] = (__bf16)src[i + e];
}

// ---------------------------------------------------------------------------
// GEMM: out[M x C] = A_bf16[M x C] @ Wb_bf16^T + bias
// MODE 0: bf16 out, [B,H,N,Dh] | MODE 1: bf16 out, [B,H,Dh,N] | MODE 2: f32 out
// Block 256 threads (8 waves); block tile 256(M) x 64(N); wave tile 32 x 64.
// ---------------------------------------------------------------------------
template<int MODE>
__global__ void gemm_kernel(const bf16_t* __restrict__ A,
                            const bf16_t* __restrict__ Wb,
                            const float* __restrict__ bias,
                            bf16_t* __restrict__ out_bf,
                            float* __restrict__ out_f,
                            int Ntok)
{
  const int K    = CDIM;
  const int wave = threadIdx.x >> 5;
  const int lane = threadIdx.x & 31;
  const int lo   = lane & 15;
  const int hi   = lane >> 4;
  const int nbase = blockIdx.x * 64;
  const int mbase = blockIdx.y * 256 + wave * 32;

  v8f acc[2][4];
#pragma unroll
  for (int mi = 0; mi < 2; ++mi)
#pragma unroll
    for (int j = 0; j < 4; ++j)
#pragma unroll
      for (int r = 0; r < 8; ++r) acc[mi][j][r] = 0.f;

  const bf16_t* arow0 = A + (size_t)(mbase + lo) * K;
  const bf16_t* arow1 = A + (size_t)(mbase + 16 + lo) * K;

  for (int kb = 0; kb < K; kb += 32) {
    v16bf a[2];
    {
      const bf16_t* ap0 = arow0 + kb + hi * 8;
      const bf16_t* ap1 = arow1 + kb + hi * 8;
#pragma unroll
      for (int e = 0; e < 8; ++e) { a[0][e] = ap0[e]; a[0][8 + e] = ap0[16 + e]; }
#pragma unroll
      for (int e = 0; e < 8; ++e) { a[1][e] = ap1[e]; a[1][8 + e] = ap1[16 + e]; }
    }
    v16bf bb[4];
#pragma unroll
    for (int j = 0; j < 4; ++j) {
      const bf16_t* wp = Wb + (size_t)(nbase + 16 * j + lo) * K + kb + hi * 8;
#pragma unroll
      for (int e = 0; e < 8; ++e) { bb[j][e] = wp[e]; bb[j][8 + e] = wp[16 + e]; }
    }
#pragma unroll
    for (int j = 0; j < 4; ++j) {
      acc[0][j] = wmma_bf16(a[0], bb[j], acc[0][j]);
      acc[1][j] = wmma_bf16(a[1], bb[j], acc[1][j]);
    }
  }

#pragma unroll
  for (int mi = 0; mi < 2; ++mi) {
#pragma unroll
    for (int j = 0; j < 4; ++j) {
      const int c  = nbase + 16 * j + lo;
      const int hh = c >> 6;
      const int dh = c & 63;
      const float bval = bias[c];
#pragma unroll
      for (int r = 0; r < 8; ++r) {
        const int R = mbase + mi * 16 + r + hi * 8;   // global row (b*Ntok + n)
        const float v = acc[mi][j][r] + bval;
        if (MODE == 2) {
          out_f[(size_t)R * CDIM + c] = v;
        } else {
          const int bb2 = R / Ntok, n = R % Ntok;
          size_t idx;
          if (MODE == 0) idx = (((size_t)(bb2 * HCONST + hh) * Ntok) + n) * DH + dh;
          else           idx = (((size_t)(bb2 * HCONST + hh) * DH) + dh) * Ntok + n;
          out_bf[idx] = (__bf16)v;
        }
      }
    }
  }
}

// ---------------------------------------------------------------------------
// RoPE2D: base=100, two 32-wide halves (y then x), 16 freqs each.
// ---------------------------------------------------------------------------
__global__ void rope_kernel(const bf16_t* __restrict__ in, bf16_t* __restrict__ out,
                            const int* __restrict__ pos, const int* __restrict__ ncp,
                            int N)
{
  const int idx   = blockIdx.x * blockDim.x + threadIdx.x;
  const int total = 2 * HCONST * N;
  if (idx >= total) return;
  const int n  = idx % N;
  const int bh = idx / N;
  const int b  = bh / HCONST;
  const int nc = *ncp;

  const bf16_t* src = in  + (size_t)bh * N * DH + (size_t)n * DH;
  bf16_t*       dst = out + (size_t)bh * N * DH + (size_t)n * DH;

  if (n < nc) {
#pragma unroll
    for (int d = 0; d < DH; ++d) dst[d] = src[d];
    return;
  }
  const float py = (float)pos[((size_t)b * N + n) * 2 + 0];
  const float px = (float)pos[((size_t)b * N + n) * 2 + 1];
#pragma unroll
  for (int half = 0; half < 2; ++half) {
    const float p    = (half == 0) ? py : px;
    const int   base = half * 32;
#pragma unroll
    for (int i = 0; i < 16; ++i) {
      const float freq = __powf(100.0f, -(float)(2 * i) / 32.0f);
      const float ang = p * freq;
      const float c = __cosf(ang), s = __sinf(ang);
      const float x1 = (float)src[base + i];
      const float x2 = (float)src[base + 16 + i];
      dst[base + i]      = (__bf16)(x1 * c - x2 * s);
      dst[base + 16 + i] = (__bf16)(x2 * c + x1 * s);
    }
  }
}

// ---------------------------------------------------------------------------
// Flash attention. Grid (Nq/128, H, B), 256 threads (8 waves).
// __launch_bounds__(256,1): give the allocator full VGPR headroom (no spills).
// K/V tiles async-DMA'd to LDS once per block, double-buffered, branch-free
// prefetch (wrapped index). Fragments preloaded in pairs; softmax reductions
// via DPP8 / DPP16 row_ror:8 butterflies.
// ---------------------------------------------------------------------------
__global__ void __launch_bounds__(256, 1)
attn_kernel(const bf16_t* __restrict__ Qr, const bf16_t* __restrict__ Kr,
            const bf16_t* __restrict__ Vt,
            const bf16_t* __restrict__ Qraw, const bf16_t* __restrict__ Kraw,
            bf16_t* __restrict__ Obuf,
            const int* __restrict__ ncp, int Nq, int Nk)
{
  __shared__ bf16_t Kbuf[2][32 * DH];   // [key][dh]   8 KB
  __shared__ bf16_t Vbuf[2][DH * 32];   // [dh][key]   8 KB
  __shared__ bf16_t pbuf[8][16 * 32];   // per-wave P staging, 8 KB

  const int tid  = threadIdx.x;
  const int wave = tid >> 5;
  const int lane = tid & 31;
  const int lo   = lane & 15;
  const int hi   = lane >> 4;
  const int b = blockIdx.z, h = blockIdx.y;
  const int qbase = blockIdx.x * 128 + wave * 16;
  const int nc = *ncp;
  const size_t bh = (size_t)(b * HCONST + h);

  const bf16_t* q    = Qr   + bh * Nq * DH;     // [Nq,64]
  const bf16_t* k    = Kr   + bh * Nk * DH;     // [Nk,64]
  const bf16_t* vt   = Vt   + bh * DH * Nk;     // [64,Nk]
  const bf16_t* qraw = Qraw + bh * Nq * DH;
  const bf16_t* kraw = Kraw + bh * Nk * DH;

  // cooperative async stage of one 32-key K tile + V tile (2 x b128 per thread)
  const int kk = tid >> 3, kc = tid & 7;         // K: 32 rows x 8 chunks
  const int vd = tid >> 2, vc = tid & 3;         // V: 64 rows x 4 chunks
  auto stage = [&](int buf, int kb) {
    async_ld_b128(&Kbuf[buf][kk * DH + kc * 8],
                  k + (size_t)(kb + kk) * DH + kc * 8);
    async_ld_b128(&Vbuf[buf][vd * 32 + vc * 8],
                  vt + (size_t)vd * Nk + kb + vc * 8);
  };

  // Q A-fragments: 16 rows x 64 k -> two 16x32 frags (loaded once per wave)
  v16bf qa[2];
  {
    const bf16_t* qrow = q + (size_t)(qbase + lo) * DH;
#pragma unroll
    for (int c2 = 0; c2 < 2; ++c2) {
      const bf16_t* ap = qrow + c2 * 32 + hi * 8;
#pragma unroll
      for (int e = 0; e < 8; ++e) { qa[c2][e] = ap[e]; qa[c2][8 + e] = ap[16 + e]; }
    }
  }

  v8f o[4];
#pragma unroll
  for (int j = 0; j < 4; ++j)
#pragma unroll
    for (int r = 0; r < 8; ++r) o[j][r] = 0.f;

  float m_i[8], l_i[8];
#pragma unroll
  for (int r = 0; r < 8; ++r) { m_i[r] = -3.0e30f; l_i[r] = 0.f; }

  const float scale = 0.125f;                   // Dh^-0.5
  bf16_t* pw = &pbuf[wave][0];

  stage(0, 0);
  wait_async0();
  __syncthreads();

  const int nblocks = Nk / 32;                  // power of two (64)
  for (int ib = 0; ib < nblocks; ++ib) {
    const int kb  = ib * 32;
    const int cur = ib & 1;
    // branch-free prefetch: last iteration re-fetches block 0 into the unread
    // buffer (harmless), keeping the loop body straight-line.
    stage(cur ^ 1, ((ib + 1) & (nblocks - 1)) * 32);

    const bf16_t* kt  = &Kbuf[cur][0];
    const bf16_t* vtl = &Vbuf[cur][0];

    // ---- S = Q*K^T: preload K frags in pairs, WMMAs interleave s0/s1
    v8f s[2];
#pragma unroll
    for (int t = 0; t < 2; ++t)
#pragma unroll
      for (int r = 0; r < 8; ++r) s[t][r] = 0.f;
#pragma unroll
    for (int c2 = 0; c2 < 2; ++c2) {
      v16bf kf[2];
#pragma unroll
      for (int t = 0; t < 2; ++t) {
        const bf16_t* bp = kt + (16 * t + lo) * DH + c2 * 32 + hi * 8;
#pragma unroll
        for (int e = 0; e < 8; ++e) { kf[t][e] = bp[e]; kf[t][8 + e] = bp[16 + e]; }
      }
      s[0] = wmma_bf16(qa[c2], kf[0], s[0]);
      s[1] = wmma_bf16(qa[c2], kf[1], s[1]);
    }

    // ---- cls fixup (uniform-guarded: only q-tile 0 or key-block 0 can hit)
    if (qbase < nc || kb < nc) {
#pragma unroll
      for (int t = 0; t < 2; ++t) {
        const int ki = kb + 16 * t + lo;
#pragma unroll
        for (int r = 0; r < 8; ++r) {
          const int qi = qbase + r + hi * 8;
          if ((qi < nc) != (ki < nc)) {
            const bf16_t* qq  = qraw + (size_t)qi * DH;
            const bf16_t* kk2 = kraw + (size_t)ki * DH;
            float accf = 0.f;
            for (int d = 0; d < DH; ++d) accf += (float)qq[d] * (float)kk2[d];
            s[t][r] = accf;
          }
        }
      }
    }
#pragma unroll
    for (int t = 0; t < 2; ++t)
#pragma unroll
      for (int r = 0; r < 8; ++r) s[t][r] *= scale;

    // ---- online softmax: DPP butterflies (VALU only)
#pragma unroll
    for (int r = 0; r < 8; ++r) {
      const float rmax = half_max16(fmaxf(s[0][r], s[1][r]));
      const float mnew = fmaxf(m_i[r], rmax);
      const float corr = __expf(m_i[r] - mnew);
      const float p0 = __expf(s[0][r] - mnew);
      const float p1 = __expf(s[1][r] - mnew);
      const float rs = half_sum16(p0 + p1);
      l_i[r] = l_i[r] * corr + rs;
      m_i[r] = mnew;
#pragma unroll
      for (int j = 0; j < 4; ++j) o[j][r] *= corr;
      const int row = r + hi * 8;               // C-layout row of this lane
      pw[row * 32 + lo]      = (__bf16)p0;
      pw[row * 32 + 16 + lo] = (__bf16)p1;
    }
    asm volatile("s_wait_dscnt 0x0" ::: "memory");   // our LDS writes -> our reads

    // ---- P as A-fragment (16x32)
    v16bf pa;
    {
      const bf16_t* prow = pw + lo * 32 + hi * 8;
#pragma unroll
      for (int e = 0; e < 8; ++e) { pa[e] = prow[e]; pa[8 + e] = prow[16 + e]; }
    }
    // ---- O += P * V: preload V frags in pairs, 4 independent accumulators
#pragma unroll
    for (int jj = 0; jj < 2; ++jj) {
      v16bf vf[2];
#pragma unroll
      for (int j2 = 0; j2 < 2; ++j2) {
        const bf16_t* vrow = vtl + (16 * (2 * jj + j2) + lo) * 32 + hi * 8;
#pragma unroll
        for (int e = 0; e < 8; ++e) { vf[j2][e] = vrow[e]; vf[j2][8 + e] = vrow[16 + e]; }
      }
      o[2 * jj]     = wmma_bf16(pa, vf[0], o[2 * jj]);
      o[2 * jj + 1] = wmma_bf16(pa, vf[1], o[2 * jj + 1]);
    }

    wait_async0();        // next tile's DMA landed
    __syncthreads();      // everyone done reading cur; next stage may overwrite
  }

  // ---- normalize and store to [B,Nq,C] bf16 (C index = h*64 + dh)
#pragma unroll
  for (int j = 0; j < 4; ++j) {
#pragma unroll
    for (int r = 0; r < 8; ++r) {
      const int qi = qbase + r + hi * 8;
      const int cc = h * DH + 16 * j + lo;
      const float val = o[j][r] / l_i[r];
      Obuf[((size_t)b * Nq + qi) * CDIM + cc] = (__bf16)val;
    }
  }
}

// ---------------------------------------------------------------------------
extern "C" void kernel_launch(void* const* d_in, const int* in_sizes, int n_in,
                              void* d_out, int out_size, void* d_ws, size_t ws_size,
                              hipStream_t stream)
{
  const float* query = (const float*)d_in[0];
  const float* key_  = (const float*)d_in[1];
  const float* value = (const float*)d_in[2];
  const float* Wq = (const float*)d_in[3];
  const float* bq = (const float*)d_in[4];
  const float* Wk = (const float*)d_in[5];
  const float* bk = (const float*)d_in[6];
  const float* Wv = (const float*)d_in[7];
  const float* bv = (const float*)d_in[8];
  const float* Wp = (const float*)d_in[9];
  const float* bp = (const float*)d_in[10];
  const int* qpos = (const int*)d_in[11];
  const int* kpos = (const int*)d_in[12];
  const int* ncp  = (const int*)d_in[13];   // device scalar; read in-kernel
  float* out = (float*)d_out;

  const int B = 2, Ntok = 2048;
  const int M = B * Ntok;                       // 4096
  const size_t MB = 1u << 20;
  char* ws = (char*)d_ws;

  bf16_t* Xq    = (bf16_t*)(ws + 0 * MB);       // 8 MB  (reused as Obuf later)
  bf16_t* Xk    = (bf16_t*)(ws + 8 * MB);
  bf16_t* Xv    = (bf16_t*)(ws + 16 * MB);
  bf16_t* Qraw  = (bf16_t*)(ws + 24 * MB);
  bf16_t* Kraw  = (bf16_t*)(ws + 32 * MB);
  bf16_t* Vt    = (bf16_t*)(ws + 40 * MB);
  bf16_t* Qrope = (bf16_t*)(ws + 48 * MB);
  bf16_t* Krope = (bf16_t*)(ws + 56 * MB);
  bf16_t* Wqb   = (bf16_t*)(ws + 64 * MB);      // 2 MB each
  bf16_t* Wkb   = (bf16_t*)(ws + 66 * MB);
  bf16_t* Wvb   = (bf16_t*)(ws + 68 * MB);
  bf16_t* Wpb   = (bf16_t*)(ws + 70 * MB);
  bf16_t* Obuf  = Xq;                           // Xq dead after Q projection

  // 1) one-time fp32 -> bf16 conversion of activations and weights
  const int nX = M * CDIM;                      // 4 Mi
  const int nW = CDIM * CDIM;                   // 1 Mi
  cvt_bf16_kernel<<<nX / (8 * 256), 256, 0, stream>>>(query, Xq, nX);
  cvt_bf16_kernel<<<nX / (8 * 256), 256, 0, stream>>>(key_,  Xk, nX);
  cvt_bf16_kernel<<<nX / (8 * 256), 256, 0, stream>>>(value, Xv, nX);
  cvt_bf16_kernel<<<nW / (8 * 256), 256, 0, stream>>>(Wq, Wqb, nW);
  cvt_bf16_kernel<<<nW / (8 * 256), 256, 0, stream>>>(Wk, Wkb, nW);
  cvt_bf16_kernel<<<nW / (8 * 256), 256, 0, stream>>>(Wv, Wvb, nW);
  cvt_bf16_kernel<<<nW / (8 * 256), 256, 0, stream>>>(Wp, Wpb, nW);

  // 2) projections
  dim3 gblk(256);
  dim3 ggrid(CDIM / 64, M / 256);
  gemm_kernel<0><<<ggrid, gblk, 0, stream>>>(Xq, Wqb, bq, Qraw, nullptr, Ntok);
  gemm_kernel<0><<<ggrid, gblk, 0, stream>>>(Xk, Wkb, bk, Kraw, nullptr, Ntok);
  gemm_kernel<1><<<ggrid, gblk, 0, stream>>>(Xv, Wvb, bv, Vt,   nullptr, Ntok);

  // 3) RoPE2D
  const int ropeThreads = B * HCONST * Ntok;
  rope_kernel<<<ropeThreads / 256, 256, 0, stream>>>(Qraw, Qrope, qpos, ncp, Ntok);
  rope_kernel<<<ropeThreads / 256, 256, 0, stream>>>(Kraw, Krope, kpos, ncp, Ntok);

  // 4) flash attention
  dim3 agrid(Ntok / 128, HCONST, B);
  attn_kernel<<<agrid, 256, 0, stream>>>(Qrope, Krope, Vt, Qraw, Kraw, Obuf,
                                         ncp, Ntok, Ntok);

  // 5) output projection (fp32 out)
  gemm_kernel<2><<<ggrid, gblk, 0, stream>>>(Obuf, Wpb, bp, nullptr, out, Ntok);
}
